// SimpleGraphEncoder_16612933501305
// MI455X (gfx1250) — compile-verified, matrix operand order checked by reading, not run on hardware
//
#include <hip/hip_runtime.h>

#define NNODES 100000
#define NEDGES 1600000
#define DD 64
#define EFF 4
#define KIN_E 68   // D + EF
#define KPAD_E 96  // padded to 3 * 32
#define KIN_N 64   // 2 * 32

typedef __attribute__((ext_vector_type(16))) __bf16 v16bf;
typedef __attribute__((ext_vector_type(8)))  __bf16 v8bf;
typedef __attribute__((ext_vector_type(8)))  float  v8f;

union Frag { v16bf v; v8bf h[2]; };

__device__ __forceinline__ __bf16 to_bf16(float f) {
    union { float f; unsigned u; } in; in.f = f;
    unsigned r = in.u + 0x7FFFu + ((in.u >> 16) & 1u);   // round-to-nearest-even
    union { unsigned short s; __bf16 b; } out;
    out.s = (unsigned short)(r >> 16);
    return out.b;
}

// ---------------------------------------------------------------------------
// x0[i] = emb[x_ids[i/64] * 64 + i%64]
// ---------------------------------------------------------------------------
__global__ __launch_bounds__(256) void gather_embed(const int* __restrict__ ids,
                                                    const float* __restrict__ emb,
                                                    float* __restrict__ x) {
    long i = (long)blockIdx.x * 256 + threadIdx.x;
    if (i < (long)NNODES * DD) {
        int n = (int)(i >> 6);
        int d = (int)(i & 63);
        x[i] = emb[(size_t)ids[n] * DD + d];
    }
}

// ---------------------------------------------------------------------------
// Edge MLP + scatter-add:  agg[dst[e]] += relu(W * [x[src[e]] || eattr[e]] + b)
// One wave per 16-edge tile, 8 tiles per block.
// ---------------------------------------------------------------------------
__global__ __launch_bounds__(256) void edge_mlp_scatter(
    const float* __restrict__ x, const int* __restrict__ src,
    const int* __restrict__ dst, const float* __restrict__ eattr,
    const float* __restrict__ W, const float* __restrict__ b,
    float* __restrict__ agg, int ntiles)
{
    __shared__ __bf16 sW[DD][KPAD_E];        // B-matrix rows: W[n][k], k padded w/ 0
    __shared__ float  sB[DD];
    __shared__ __bf16 sA[8][16][KPAD_E];     // per-wave A staging
    __shared__ int    sDst[8][16];

    const int tid  = threadIdx.x;
    const int wid  = tid >> 5;
    const int lane = tid & 31;

    for (int i = tid; i < DD * KPAD_E; i += 256) {
        int n = i / KPAD_E, k = i % KPAD_E;
        sW[n][k] = (k < KIN_E) ? to_bf16(W[n * KIN_E + k]) : to_bf16(0.0f);
    }
    if (tid < DD) sB[tid] = b[tid];

    const int tile   = blockIdx.x * 8 + wid;
    const bool active = tile < ntiles;
    const int ebase  = tile * 16;

    if (active) {
        for (int e = 0; e < 16; ++e) {
            int edge = ebase + e;
            const float* xr = x + (size_t)src[edge] * DD;
            sA[wid][e][2 * lane]     = to_bf16(xr[2 * lane]);
            sA[wid][e][2 * lane + 1] = to_bf16(xr[2 * lane + 1]);
            // lanes 0..3: edge features at k=64..67; lanes 4..31: zero pad k=68..95
            sA[wid][e][DD + lane] = (lane < EFF)
                ? to_bf16(eattr[(size_t)edge * EFF + lane]) : to_bf16(0.0f);
            if (lane == 0) sDst[wid][e] = dst[edge];
        }
    }
    __syncthreads();
    if (!active) return;

    const int mrow  = lane & 15;
    const int khalf = lane >> 4;

    v8f acc[4];
    #pragma unroll
    for (int nt = 0; nt < 4; ++nt) {
        float bias = sB[nt * 16 + mrow];
        #pragma unroll
        for (int v = 0; v < 8; ++v) acc[nt][v] = bias;
    }

    #pragma unroll
    for (int kc = 0; kc < 3; ++kc) {
        const int kb = kc * 32 + khalf * 8;
        Frag a;
        const __bf16* rowA = &sA[wid][mrow][0];
        a.h[0] = *(const v8bf*)(rowA + kb);
        a.h[1] = *(const v8bf*)(rowA + kb + 16);
        #pragma unroll
        for (int nt = 0; nt < 4; ++nt) {
            Frag bw;
            const __bf16* rowB = &sW[nt * 16 + mrow][0];
            bw.h[0] = *(const v8bf*)(rowB + kb);
            bw.h[1] = *(const v8bf*)(rowB + kb + 16);
            acc[nt] = __builtin_amdgcn_wmma_f32_16x16x32_bf16(
                false, a.v, false, bw.v, (short)0, acc[nt], false, false);
        }
    }

    #pragma unroll
    for (int nt = 0; nt < 4; ++nt) {
        const int n = nt * 16 + mrow;
        #pragma unroll
        for (int v = 0; v < 8; ++v) {
            const int m = v + khalf * 8;           // edge row in tile
            float val = acc[nt][v];
            val = val > 0.0f ? val : 0.0f;
            atomicAdd(&agg[(size_t)sDst[wid][m] * DD + n], val);
        }
    }
}

// ---------------------------------------------------------------------------
// Node update:  xout = relu(Wu * (x + agg) + bu)
// ---------------------------------------------------------------------------
__global__ __launch_bounds__(256) void node_update(
    const float* __restrict__ x, const float* __restrict__ agg,
    const float* __restrict__ Wu, const float* __restrict__ bu,
    float* __restrict__ xout, int ntiles)
{
    __shared__ __bf16 sW[DD][KIN_N];
    __shared__ float  sB[DD];
    __shared__ __bf16 sA[8][16][KIN_N];

    const int tid  = threadIdx.x;
    const int wid  = tid >> 5;
    const int lane = tid & 31;

    for (int i = tid; i < DD * KIN_N; i += 256)
        sW[i / KIN_N][i % KIN_N] = to_bf16(Wu[i]);
    if (tid < DD) sB[tid] = bu[tid];

    const int tile    = blockIdx.x * 8 + wid;
    const bool active = tile < ntiles;
    const int nbase   = tile * 16;

    if (active) {
        for (int e = 0; e < 16; ++e) {
            size_t row = (size_t)(nbase + e) * DD;
            sA[wid][e][2 * lane] =
                to_bf16(x[row + 2 * lane] + agg[row + 2 * lane]);
            sA[wid][e][2 * lane + 1] =
                to_bf16(x[row + 2 * lane + 1] + agg[row + 2 * lane + 1]);
        }
    }
    __syncthreads();
    if (!active) return;

    const int mrow  = lane & 15;
    const int khalf = lane >> 4;

    v8f acc[4];
    #pragma unroll
    for (int nt = 0; nt < 4; ++nt) {
        float bias = sB[nt * 16 + mrow];
        #pragma unroll
        for (int v = 0; v < 8; ++v) acc[nt][v] = bias;
    }

    #pragma unroll
    for (int kc = 0; kc < 2; ++kc) {
        const int kb = kc * 32 + khalf * 8;
        Frag a;
        const __bf16* rowA = &sA[wid][mrow][0];
        a.h[0] = *(const v8bf*)(rowA + kb);
        a.h[1] = *(const v8bf*)(rowA + kb + 16);
        #pragma unroll
        for (int nt = 0; nt < 4; ++nt) {
            Frag bw;
            const __bf16* rowB = &sW[nt * 16 + mrow][0];
            bw.h[0] = *(const v8bf*)(rowB + kb);
            bw.h[1] = *(const v8bf*)(rowB + kb + 16);
            acc[nt] = __builtin_amdgcn_wmma_f32_16x16x32_bf16(
                false, a.v, false, bw.v, (short)0, acc[nt], false, false);
        }
    }

    #pragma unroll
    for (int nt = 0; nt < 4; ++nt) {
        const int n = nt * 16 + mrow;
        #pragma unroll
        for (int v = 0; v < 8; ++v) {
            const int m = v + khalf * 8;
            float val = acc[nt][v];
            val = val > 0.0f ? val : 0.0f;
            xout[(size_t)(nbase + m) * DD + n] = val;
        }
    }
}

// ---------------------------------------------------------------------------
// out[c] = mean over nodes of x[:, c]
// ---------------------------------------------------------------------------
__global__ __launch_bounds__(256) void mean_reduce(const float* __restrict__ x,
                                                   float* __restrict__ out) {
    const int c = threadIdx.x & 63;
    const int g = threadIdx.x >> 6;     // 0..3
    float sum = 0.0f;
    for (int n = blockIdx.x * 4 + g; n < NNODES; n += gridDim.x * 4)
        sum += x[(size_t)n * DD + c];
    __shared__ float red[4][DD];
    red[g][c] = sum;
    __syncthreads();
    if (g == 0) {
        float t = red[0][c] + red[1][c] + red[2][c] + red[3][c];
        atomicAdd(&out[c], t * (1.0f / (float)NNODES));
    }
}

extern "C" void kernel_launch(void* const* d_in, const int* in_sizes, int n_in,
                              void* d_out, int out_size, void* d_ws, size_t ws_size,
                              hipStream_t stream) {
    (void)in_sizes; (void)n_in; (void)out_size; (void)ws_size;
    const int*   x_ids = (const int*)d_in[0];
    const int*   eidx  = (const int*)d_in[1];
    const float* eattr = (const float*)d_in[2];
    const float* emb   = (const float*)d_in[3];
    const float* W1    = (const float*)d_in[4];
    const float* b1    = (const float*)d_in[5];
    const float* Wu1   = (const float*)d_in[6];
    const float* bu1   = (const float*)d_in[7];
    const float* W2    = (const float*)d_in[8];
    const float* b2    = (const float*)d_in[9];
    const float* Wu2   = (const float*)d_in[10];
    const float* bu2   = (const float*)d_in[11];

    const int* src = eidx;            // edge_index[0]
    const int* dst = eidx + NEDGES;   // edge_index[1]
    float* out = (float*)d_out;

    float* x0  = (float*)d_ws;
    float* x1  = x0 + (size_t)NNODES * DD;
    float* agg = x1 + (size_t)NNODES * DD;
    const size_t featBytes = (size_t)NNODES * DD * sizeof(float);

    const int etiles  = NEDGES / 16;            // 100000
    const int eblocks = (etiles + 7) / 8;       // 12500
    const int ntiles  = NNODES / 16;            // 6250
    const int nblocks = (ntiles + 7) / 8;       // 782

    gather_embed<<<(NNODES * DD + 255) / 256, 256, 0, stream>>>(x_ids, emb, x0);

    // ---- round 1 ----
    hipMemsetAsync(agg, 0, featBytes, stream);
    edge_mlp_scatter<<<eblocks, 256, 0, stream>>>(x0, src, dst, eattr, W1, b1, agg, etiles);
    node_update<<<nblocks, 256, 0, stream>>>(x0, agg, Wu1, bu1, x1, ntiles);

    // ---- round 2 ----
    hipMemsetAsync(agg, 0, featBytes, stream);
    edge_mlp_scatter<<<eblocks, 256, 0, stream>>>(x1, src, dst, eattr, W2, b2, agg, etiles);
    node_update<<<nblocks, 256, 0, stream>>>(x1, agg, Wu2, bu2, x0, ntiles);

    // ---- mean over nodes ----
    hipMemsetAsync(out, 0, DD * sizeof(float), stream);
    mean_reduce<<<256, 256, 0, stream>>>(x0, out);
}
